// BidirectionalAttention_62826781605919
// MI455X (gfx1250) — compile-verified, matrix operand order checked by reading, not run on hardware
//
#include <hip/hip_runtime.h>
#include <stdint.h>

typedef __attribute__((ext_vector_type(16))) _Float16 v16h;
typedef __attribute__((ext_vector_type(8)))  _Float16 v8h;
typedef __attribute__((ext_vector_type(8)))  float    v8f;

#define WMMA_F16(a, b, c) \
  __builtin_amdgcn_wmma_f32_16x16x32_f16(false, (a), false, (b), (short)0, (c), false, false)

static __device__ __forceinline__ v8f v8f_zero() {
  v8f z = {0.f, 0.f, 0.f, 0.f, 0.f, 0.f, 0.f, 0.f};
  return z;
}

// ---------------- CDNA5 async copy to LDS (ASYNCcnt-tracked) ----------------
// LDS byte offset of a __shared__ object == low 32 bits of its generic pointer
// (flat aperture, CDNA5 ISA).
static __device__ __forceinline__ unsigned lds_off(const void* p) {
  return (unsigned)(uintptr_t)p;
}
static __device__ __forceinline__ void async_copy_b128(unsigned lds_byte, const void* gptr) {
  asm volatile("global_load_async_to_lds_b128 %0, %1, off"
               :: "v"(lds_byte), "v"(gptr) : "memory");
}
static __device__ __forceinline__ void wait_async_le8() {  // older batch complete
  asm volatile("s_wait_asynccnt 0x8" ::: "memory");
}
static __device__ __forceinline__ void wait_async_le0() {
  asm volatile("s_wait_asynccnt 0x0" ::: "memory");
}

// ---------------- WMMA fragment loads ----------------
// 16x32 f16 operand (CDNA5 ISA 7.12.2): lane l holds row l%16,
// K halves (l/16)*8 + {0..7} and + {16..23}: two 16-byte loads per lane.
static __device__ __forceinline__ v16h load_frag(const _Float16* __restrict__ base,
                                                 int row0, int ld, int k0) {
  const int lane = threadIdx.x & 31;
  const int r  = row0 + (lane & 15);
  const int kb = k0 + ((lane >> 4) << 3);
  const _Float16* p = base + (size_t)r * ld + kb;
  v8h lo = *(const v8h*)(p);
  v8h hi = *(const v8h*)(p + 16);
  v16h out;
#pragma unroll
  for (int i = 0; i < 8; ++i) { out[i] = lo[i]; out[i + 8] = hi[i]; }
  return out;
}

static __device__ __forceinline__ v16h load_frag_lds(const _Float16* base,
                                                     int row0, int ld, int k0) {
  const int lane = threadIdx.x & 31;
  const int r  = row0 + (lane & 15);
  const int kb = k0 + ((lane >> 4) << 3);
  const _Float16* p = base + r * ld + kb;
  v8h lo = *(const v8h*)(p);   // ds_load_b128
  v8h hi = *(const v8h*)(p + 16);
  v16h out;
#pragma unroll
  for (int i = 0; i < 8; ++i) { out[i] = lo[i]; out[i + 8] = hi[i]; }
  return out;
}

// ---------------- conversion / transpose kernels ----------------

__global__ __launch_bounds__(256) void cvt_kernel(const float* __restrict__ in,
                                                  _Float16* __restrict__ out, int n) {
  for (int i = blockIdx.x * 256 + threadIdx.x; i < n; i += gridDim.x * 256)
    out[i] = (_Float16)in[i];
}

// W [K, A] f32 -> Wt [A, K] f16
__global__ __launch_bounds__(256) void wt_kernel(const float* __restrict__ W,
                                                 _Float16* __restrict__ Wt,
                                                 int K, int A) {
  int i = blockIdx.x * 256 + threadIdx.x;
  if (i < K * A) {
    int k = i / A, n = i % A;
    Wt[(size_t)n * K + k] = (_Float16)W[i];
  }
}

// v [B, M, D] f32 -> vt [B, D, M] f16 (LDS-tiled transpose)
__global__ __launch_bounds__(256) void vtrans_kernel(const float* __restrict__ v,
                                                     _Float16* __restrict__ vt,
                                                     int M, int D) {
  __shared__ __align__(16) _Float16 tile[32][33];
  const int b = blockIdx.z;
  const int m0 = blockIdx.x * 32, d0 = blockIdx.y * 32;
  const int tx = threadIdx.x & 31, ty = threadIdx.x >> 5;
#pragma unroll
  for (int r = ty; r < 32; r += 8)
    tile[r][tx] = (_Float16)v[((size_t)b * M + m0 + r) * D + d0 + tx];
  __syncthreads();
#pragma unroll
  for (int r = ty; r < 32; r += 8)
    vt[((size_t)b * D + d0 + r) * M + m0 + tx] = tile[tx][r];
}

// ---------------- projection: kp = k @ W + bias (f16 out) ----------------
// Wt staged once per block into LDS via async copies; per-wave A frags from
// global, B frags from LDS. Block = 8 waves x 16 rows = 128 rows, 128 cols.

__global__ __launch_bounds__(256) void proj_kernel(const _Float16* __restrict__ kh,
                                                   const _Float16* __restrict__ Wt,
                                                   const float* __restrict__ bias,
                                                   _Float16* __restrict__ kp,
                                                   int N, int K, int A) {
  constexpr int LDW = 264;                 // 256 + 8 halves: bank stagger, 16B aligned
  __shared__ __align__(16) _Float16 lds_w[128 * LDW];

  const int b = blockIdx.y;
  const int row0 = blockIdx.x * 128;
  const int tid = threadIdx.x, wid = tid >> 5, lane = tid & 31;
  const _Float16* kb = kh + (size_t)b * N * K;
  _Float16* ob = kp + (size_t)b * N * A;
  const int wrow = row0 + wid * 16;

  {  // stage Wt [128 rows x 256 halves]: 4096 16B chunks, 16 per thread
    unsigned base = lds_off(lds_w);
#pragma unroll
    for (int p = 0; p < 16; ++p) {
      int c = tid + p * 256;
      int row = c >> 5, col8 = (c & 31) << 3;
      async_copy_b128(base + (unsigned)(row * LDW + col8) * 2,
                      Wt + (size_t)row * K + col8);
    }
    wait_async_le0();
    __syncthreads();
  }

  v8f acc[8];
#pragma unroll
  for (int i = 0; i < 8; ++i) acc[i] = v8f_zero();

  for (int k0 = 0; k0 < K; k0 += 32) {
    v16h a = load_frag(kb, wrow, K, k0);
#pragma unroll
    for (int ns = 0; ns < 8; ++ns) {
      v16h bf = load_frag_lds(lds_w, ns * 16, LDW, k0);
      acc[ns] = WMMA_F16(a, bf, acc[ns]);
    }
  }

  const int half = lane >> 4, c = lane & 15;
  const int rbase = wrow + half * 8;
#pragma unroll
  for (int ns = 0; ns < 8; ++ns) {
    float bv = bias[ns * 16 + c];
#pragma unroll
    for (int v = 0; v < 8; ++v)
      ob[(size_t)(rbase + v) * A + ns * 16 + c] = (_Float16)(acc[ns][v] + bv);
  }
}

// ---------------- attention core ----------------
// S' = kpA @ kpB^T; P = row-softmax(S'); O = P @ V (V transposed [D, N]).
// A-tile (64x128) resident in VGPRs; B-tiles double-buffered in LDS via
// async copies; online (m, s) stats; O accumulators resident across j.

constexpr int ATM = 64, ATJ = 128, LDB = 136, LDP = 136;

static __device__ __forceinline__ void stage_b_tile(const _Float16* __restrict__ Bb,
                                                    int j0, int A, _Float16* buf) {
  const int tid = threadIdx.x;
  unsigned base = lds_off(buf);
#pragma unroll
  for (int p = 0; p < 8; ++p) {          // 2048 16B chunks, 8 per thread
    int c = tid + p * 256;
    int row = c >> 4, col8 = (c & 15) << 3;
    async_copy_b128(base + (unsigned)(row * LDB + col8) * 2,
                    Bb + (size_t)(j0 + row) * A + col8);
  }
}

static __device__ __forceinline__ void score_tile_lds(const _Float16* bt,
                                                      const v16h (&aF)[2][4], int sc,
                                                      v8f& s00, v8f& s01, v8f& s10, v8f& s11) {
  s00 = v8f_zero(); s01 = v8f_zero(); s10 = v8f_zero(); s11 = v8f_zero();
#pragma unroll
  for (int kk = 0; kk < 4; ++kk) {
    v16h b0 = load_frag_lds(bt, sc * 32,      LDB, kk * 32);
    v16h b1 = load_frag_lds(bt, sc * 32 + 16, LDB, kk * 32);
    s00 = WMMA_F16(aF[0][kk], b0, s00); s01 = WMMA_F16(aF[0][kk], b1, s01);
    s10 = WMMA_F16(aF[1][kk], b0, s10); s11 = WMMA_F16(aF[1][kk], b1, s11);
  }
}

__global__ __launch_bounds__(256) void attn_kernel(const _Float16* __restrict__ kpA,
                                                   const _Float16* __restrict__ kpB,
                                                   const _Float16* __restrict__ vt,
                                                   float* __restrict__ out,
                                                   int N, int A, int D) {
  __shared__ __align__(16) _Float16 lds_b[2][128 * LDB];
  __shared__ __align__(16) _Float16 lds_p[ATM * LDP];
  __shared__ __align__(16) float part_m[4][ATM];
  __shared__ __align__(16) float part_s[4][ATM];
  __shared__ __align__(16) float rmaxF[ATM];
  __shared__ __align__(16) float rinvF[ATM];

  const int b = blockIdx.y;
  const int row0 = blockIdx.x * ATM;
  const _Float16* Ab = kpA + (size_t)b * N * A;
  const _Float16* Bb = kpB + (size_t)b * N * A;
  const _Float16* Vb = vt + (size_t)b * D * N;
  float* Ob = out + (size_t)b * N * D;

  const int tid = threadIdx.x, lane = tid & 31, wid = tid >> 5;
  const int sr = wid >> 2, sc = wid & 3;
  const int srow = row0 + sr * 32;
  const int half = lane >> 4, col16 = lane & 15;
  const int ni = N / ATJ;

  // Persistent A fragments: 32 rows x 128 K per wave (8 x v16h = 64 VGPRs).
  v16h aF[2][4];
#pragma unroll
  for (int m = 0; m < 2; ++m)
#pragma unroll
    for (int kk = 0; kk < 4; ++kk)
      aF[m][kk] = load_frag(Ab, srow + m * 16, A, kk * 32);

  v8f s00, s01, s10, s11;

  // ---- sweep A: online row max + expsum ----
  float mrun[16], srun[16];
#pragma unroll
  for (int i = 0; i < 16; ++i) { mrun[i] = -3.0e38f; srun[i] = 0.f; }

  stage_b_tile(Bb, 0, A, lds_b[0]);
  for (int i = 0; i < ni; ++i) {
    if (i + 1 < ni) { stage_b_tile(Bb, (i + 1) * ATJ, A, lds_b[(i + 1) & 1]); wait_async_le8(); }
    else            { wait_async_le0(); }
    __syncthreads();
    score_tile_lds(lds_b[i & 1], aF, sc, s00, s01, s10, s11);
#pragma unroll
    for (int v = 0; v < 8; ++v) {
      float nm0 = fmaxf(mrun[v], fmaxf(s00[v], s01[v]));
      srun[v] = srun[v] * __expf(mrun[v] - nm0)
              + __expf(s00[v] - nm0) + __expf(s01[v] - nm0);
      mrun[v] = nm0;
      float nm1 = fmaxf(mrun[8 + v], fmaxf(s10[v], s11[v]));
      srun[8 + v] = srun[8 + v] * __expf(mrun[8 + v] - nm1)
                  + __expf(s10[v] - nm1) + __expf(s11[v] - nm1);
      mrun[8 + v] = nm1;
    }
    __syncthreads();
  }
  // merge (m, s) across the 16 lanes sharing the same rows
#pragma unroll
  for (int off = 1; off < 16; off <<= 1) {
#pragma unroll
    for (int i = 0; i < 16; ++i) {
      float mo = __shfl_xor(mrun[i], off, 32);
      float so = __shfl_xor(srun[i], off, 32);
      float nm = fmaxf(mrun[i], mo);
      srun[i] = srun[i] * __expf(mrun[i] - nm) + so * __expf(mo - nm);
      mrun[i] = nm;
    }
  }
  if (col16 == 0) {
#pragma unroll
    for (int m = 0; m < 2; ++m)
#pragma unroll
      for (int v = 0; v < 8; ++v) {
        int lr = sr * 32 + m * 16 + half * 8 + v;
        part_m[sc][lr] = mrun[m * 8 + v];
        part_s[sc][lr] = srun[m * 8 + v];
      }
  }
  __syncthreads();
  if (tid < ATM) {
    float M = fmaxf(fmaxf(part_m[0][tid], part_m[1][tid]),
                    fmaxf(part_m[2][tid], part_m[3][tid]));
    float S = part_s[0][tid] * __expf(part_m[0][tid] - M)
            + part_s[1][tid] * __expf(part_m[1][tid] - M)
            + part_s[2][tid] * __expf(part_m[2][tid] - M)
            + part_s[3][tid] * __expf(part_m[3][tid] - M);
    rmaxF[tid] = M;
    rinvF[tid] = 1.0f / S;
  }
  __syncthreads();

  // ---- sweep B: O = P @ V with resident accumulators ----
  v8f o[2][4];
#pragma unroll
  for (int m = 0; m < 2; ++m)
#pragma unroll
    for (int n = 0; n < 4; ++n) o[m][n] = v8f_zero();

  stage_b_tile(Bb, 0, A, lds_b[0]);
  for (int i = 0; i < ni; ++i) {
    const int j0 = i * ATJ;
    if (i + 1 < ni) { stage_b_tile(Bb, (i + 1) * ATJ, A, lds_b[(i + 1) & 1]); wait_async_le8(); }
    else            { wait_async_le0(); }
    __syncthreads();
    score_tile_lds(lds_b[i & 1], aF, sc, s00, s01, s10, s11);
#pragma unroll
    for (int v = 0; v < 8; ++v) {
      int lr0 = sr * 32 + half * 8 + v;
      int lr1 = lr0 + 16;
      int c0 = sc * 32 + col16;
      lds_p[lr0 * LDP + c0]      = (_Float16)(__expf(s00[v] - rmaxF[lr0]) * rinvF[lr0]);
      lds_p[lr0 * LDP + c0 + 16] = (_Float16)(__expf(s01[v] - rmaxF[lr0]) * rinvF[lr0]);
      lds_p[lr1 * LDP + c0]      = (_Float16)(__expf(s10[v] - rmaxF[lr1]) * rinvF[lr1]);
      lds_p[lr1 * LDP + c0 + 16] = (_Float16)(__expf(s11[v] - rmaxF[lr1]) * rinvF[lr1]);
    }
    __syncthreads();
#pragma unroll
    for (int k0 = 0; k0 < ATJ; k0 += 32) {
      v16h p0 = load_frag_lds(lds_p, sr * 32,      LDP, k0);
      v16h p1 = load_frag_lds(lds_p, sr * 32 + 16, LDP, k0);
#pragma unroll
      for (int n = 0; n < 4; ++n) {
        v16h bf = load_frag(Vb, sc * 64 + n * 16, N, j0 + k0);
        o[0][n] = WMMA_F16(p0, bf, o[0][n]);
        o[1][n] = WMMA_F16(p1, bf, o[1][n]);
      }
    }
    __syncthreads();
  }

#pragma unroll
  for (int m = 0; m < 2; ++m)
#pragma unroll
    for (int n = 0; n < 4; ++n)
#pragma unroll
      for (int v = 0; v < 8; ++v) {
        int r = row0 + sr * 32 + m * 16 + half * 8 + v;
        int c = sc * 64 + n * 16 + col16;
        Ob[(size_t)r * D + c] = o[m][n][v];
      }
}

// ---------------- host ----------------

extern "C" void kernel_launch(void* const* d_in, const int* in_sizes, int n_in,
                              void* d_out, int out_size, void* d_ws, size_t ws_size,
                              hipStream_t stream) {
  (void)in_sizes; (void)n_in; (void)out_size; (void)ws_size;
  const int B = 8, N = 2048, K = 256, A = 128, D = 256;

  const float* k1 = (const float*)d_in[0];
  const float* k2 = (const float*)d_in[1];
  const float* v1 = (const float*)d_in[2];
  const float* v2 = (const float*)d_in[3];
  const float* W1 = (const float*)d_in[4];
  const float* b1 = (const float*)d_in[5];
  const float* W2 = (const float*)d_in[6];
  const float* b2 = (const float*)d_in[7];
  float* out = (float*)d_out;

  char* ws = (char*)d_ws;
  size_t off = 0;
  auto alloc = [&](size_t bytes) -> char* {
    char* p = ws + off;
    off += (bytes + 255) & ~(size_t)255;
    return p;
  };
  _Float16* k1h = (_Float16*)alloc((size_t)B * N * K * 2);
  _Float16* k2h = (_Float16*)alloc((size_t)B * N * K * 2);
  _Float16* W1t = (_Float16*)alloc((size_t)K * A * 2);
  _Float16* W2t = (_Float16*)alloc((size_t)K * A * 2);
  _Float16* v1t = (_Float16*)alloc((size_t)B * D * N * 2);
  _Float16* v2t = (_Float16*)alloc((size_t)B * D * N * 2);
  _Float16* k1p = (_Float16*)alloc((size_t)B * N * A * 2);
  _Float16* k2p = (_Float16*)alloc((size_t)B * N * A * 2);

  const int nk = B * N * K;
  cvt_kernel<<<1024, 256, 0, stream>>>(k1, k1h, nk);
  cvt_kernel<<<1024, 256, 0, stream>>>(k2, k2h, nk);
  wt_kernel<<<(K * A + 255) / 256, 256, 0, stream>>>(W1, W1t, K, A);
  wt_kernel<<<(K * A + 255) / 256, 256, 0, stream>>>(W2, W2t, K, A);
  dim3 tg(N / 32, D / 32, B);
  vtrans_kernel<<<tg, 256, 0, stream>>>(v1, v1t, N, D);
  vtrans_kernel<<<tg, 256, 0, stream>>>(v2, v2t, N, D);
  dim3 pg(N / 128, B);
  proj_kernel<<<pg, 256, 0, stream>>>(k1h, W1t, b1, k1p, N, K, A);
  proj_kernel<<<pg, 256, 0, stream>>>(k2h, W2t, b2, k2p, N, K, A);

  dim3 ag(N / 64, B);
  const size_t half = (size_t)B * N * D;
  // o2 = row-softmax(k1p @ k2p^T) @ v2
  attn_kernel<<<ag, 256, 0, stream>>>(k1p, k2p, v2t, out, N, A, D);
  // o1 = row-softmax(k2p @ k1p^T) @ v1  (column-softmax path, transposed)
  attn_kernel<<<ag, 256, 0, stream>>>(k2p, k1p, v1t, out + half, N, A, D);
}
